// EdgeEncoder_75617194213671
// MI455X (gfx1250) — matrix operand order repaired and checked by reading, not compiled, not run on hardware
//
#include <hip/hip_runtime.h>

// ---------------------------------------------------------------------------
// EdgeEncoder for MI455X (gfx1250): WMMA flash-attention over edges.
// E=8192 edges, N=2048 nodes, D_EDGE=64, D_NODE=128, EMBED=32, HEADS=4 (hd=16)
// Round 4: double-buffered async (global_load_async_to_lds_b128 / ASYNCcnt)
// pipeline in the attention key loop; DMA of chunk t+1 overlaps compute of t.
// ---------------------------------------------------------------------------

#define E_EDGES 8192
#define D_EDGE  64
#define D_NODE  128
#define EMBED   32
#define HEADS   4
#define HD      16

typedef _Float16 half_t;
typedef __attribute__((ext_vector_type(16))) _Float16 v16h;
typedef __attribute__((ext_vector_type(8)))  _Float16 v8h;
typedef __attribute__((ext_vector_type(8)))  float    v8f;
typedef __attribute__((ext_vector_type(4)))  int      v4i;

// -------------------- optional CDNA5 async global->LDS ---------------------
#if defined(__has_builtin)
#if __has_builtin(__builtin_amdgcn_global_load_async_to_lds_b128)
#define HAVE_ASYNC_LDS 1
#endif
#endif
#ifndef HAVE_ASYNC_LDS
#define HAVE_ASYNC_LDS 0
#endif

#if HAVE_ASYNC_LDS
typedef __attribute__((address_space(1))) v4i gv4i_t;   // global (AS1) int4
typedef __attribute__((address_space(3))) v4i lv4i_t;   // LDS    (AS3) int4
__device__ __forceinline__ void async_copy16(const void* gsrc, void* ldst) {
    // AS1 pointer: full 64-bit flat address (global aperture is identity).
    // AS3 pointer: low 32 bits of flat address are the LDS byte offset.
    gv4i_t* g = (gv4i_t*)(unsigned long long)(size_t)gsrc;
    lv4i_t* l = (lv4i_t*)(unsigned int)(size_t)ldst;
    __builtin_amdgcn_global_load_async_to_lds_b128(g, l, 0, 0);
}
__device__ __forceinline__ void async_wait0() {
#if __has_builtin(__builtin_amdgcn_s_wait_asynccnt)
    __builtin_amdgcn_s_wait_asynccnt(0);
#else
    asm volatile("s_wait_asynccnt 0x0" ::: "memory");
#endif
}
#endif

// ---------------------------------------------------------------------------
__device__ __forceinline__ v16h zero_v16h() {
    v16h z;
#pragma unroll
    for (int i = 0; i < 16; ++i) z[i] = (half_t)0.0f;
    return z;
}
__device__ __forceinline__ v8f zero_v8f() {
    v8f z;
#pragma unroll
    for (int i = 0; i < 8; ++i) z[i] = 0.0f;
    return z;
}

// A-fragment (16x32 f16) from row-major [16][ld]: per-lane one contiguous
// 32-byte run (lane<16 -> M=lane, K=0..15; lane>=16 ->
// M=lane-16, K=16..31).
__device__ __forceinline__ v16h load_A(const half_t* base, int ld, int koff, int lane) {
    int m = lane & 15, hi = lane >> 4;
    const half_t* row = base + m * ld + koff + hi * 16;
    v8h lo = *(const v8h*)(row);
    v8h hh = *(const v8h*)(row + 8);
    v16h a;
#pragma unroll
    for (int j = 0; j < 8; ++j) { a[j] = lo[j]; a[j + 8] = hh[j]; }
    return a;
}

// A-fragment with only K=0..15 valid (head_dim 16, upper K zero-padded).
__device__ __forceinline__ v16h load_A_k16(const half_t* base, int ld, int lane) {
    int m = lane & 15, hi = lane >> 4;
    v16h a = zero_v16h();
    if (!hi) {
        v8h lo = *(const v8h*)(base + m * ld);
        v8h hh = *(const v8h*)(base + m * ld + 8);
#pragma unroll
        for (int j = 0; j < 8; ++j) { a[j] = lo[j]; a[j + 8] = hh[j]; }
    }
    return a;
}

// B-fragment from pre-swizzled LDS buffer: [frag][lane][16 halfs] contiguous.
__device__ __forceinline__ v16h load_B_frag(const half_t* fragbuf, int frag, int lane) {
    return *(const v16h*)(fragbuf + (size_t)(frag * 32 + lane) * 16);
}

// B-fragment for QK^T: B[d][n] = keytile[n*ld + d], d=0..15 valid; per-lane
// contiguous 32-byte read (upper-half lanes all zero, d=16..31 padding).
__device__ __forceinline__ v16h load_B_kT16(const half_t* keytile, int ld, int lane) {
    int n = lane & 15, hi = lane >> 4;
    v16h b;
    if (!hi) b = *(const v16h*)(keytile + n * ld);
    else     b = zero_v16h();
    return b;
}

// B-fragment for P@V from transposed V chunk VsT[64 dims][64 keys]:
// lane element j = V[key kbase + (lane>>4)*16 + j][dim hd0 + (lane&15)] -> contiguous.
__device__ __forceinline__ v16h load_B_vT(const half_t* VsT, int kbase, int hd0, int lane) {
    int n = lane & 15, hi = lane >> 4;
    return *(const v16h*)(VsT + (hd0 + n) * 64 + kbase + hi * 16);
}

__device__ __forceinline__ v8f wmma_f16(v16h a, v16h b, v8f c) {
    return __builtin_amdgcn_wmma_f32_16x16x32_f16(false, a, false, b, (short)0, c,
                                                  false, false);
}

// 16-lane reductions (lane halves hold disjoint C-matrix row groups; wave32).
__device__ __forceinline__ float red_max16(float x) {
#pragma unroll
    for (int m = 1; m <= 8; m <<= 1) x = fmaxf(x, __shfl_xor(x, m, 32));
    return x;
}
__device__ __forceinline__ float red_sum16(float x) {
#pragma unroll
    for (int m = 1; m <= 8; m <<= 1) x += __shfl_xor(x, m, 32);
    return x;
}

// Stage an f32 weight (Kdim x Ndim, row-major) into LDS in B-fragment order:
// dst[i] with i = ((kk*GT+g)*32 + lane)*16 + j holds W[(kk*32+(lane>>4)*16+j)*Ndim + n].
__device__ __forceinline__ void stage_B_swizzled(half_t* dst, const float* W,
                                                 int Ndim, int GT, int total,
                                                 int tid, int nthr) {
    for (int i = tid; i < total; i += nthr) {
        int f  = i >> 9;
        int ln = (i >> 4) & 31;
        int j  = i & 15;
        int kk = f / GT, g = f - kk * GT;
        int n  = g * 16 + (ln & 15);
        int k  = kk * 32 + ((ln >> 4) ? 16 : 0) + j;
        dst[i] = (half_t)W[(size_t)k * Ndim + n];
    }
}

// ---------------------------------------------------------------------------
// Kernel 1: gather + h = edge_feats + (nf[src]+nf[dst])@Wn ; q,k,v = h@W{q,k,v}
// v is emitted TRANSPOSED (vT[dim][edge]) for the attention PV matmul.
// 128 threads = 4 waves; each wave owns a 16-edge tile. grid = E/64 = 128.
// ---------------------------------------------------------------------------
__global__ void __launch_bounds__(128) prep_kernel(
    const float* __restrict__ edge_feats, const float* __restrict__ node_feats,
    const float* __restrict__ Wn, const float* __restrict__ Wq,
    const float* __restrict__ Wk, const float* __restrict__ Wv,
    const int* __restrict__ edge_index,
    half_t* __restrict__ qbuf, half_t* __restrict__ kbuf, half_t* __restrict__ vtbuf)
{
    __shared__ __align__(32) half_t WnF[16 * 512];          // 16 KB, 4x4 frags
    __shared__ __align__(32) half_t WF[8 * 512];            //  8 KB, 2x4 frags (Wq/Wk/Wv)
    __shared__ __align__(32) half_t xsum[4][16 * D_NODE];   // 16 KB
    __shared__ __align__(32) half_t hbuf[4][16 * D_EDGE];   //  8 KB

    const int tid  = threadIdx.x;
    const int lane = tid & 31;
    const int wv   = tid >> 5;
    const int e0   = (blockIdx.x * 4 + wv) * 16;
    const int col  = lane & 15, hi = lane >> 4;

    stage_B_swizzled(WnF, Wn, D_EDGE, 4, 16 * 512, tid, 128);

    const int* srci = edge_index;
    const int* dsti = edge_index + E_EDGES;

    // Irregular gather: per-edge endpoint sum, converted to f16 in LDS.
    for (int i = lane; i < 16 * D_NODE; i += 32) {
        int r = i >> 7, d = i & 127;
        int e = e0 + r;
        float s = node_feats[(size_t)srci[e] * D_NODE + d] +
                  node_feats[(size_t)dsti[e] * D_NODE + d];
        xsum[wv][r * D_NODE + d] = (half_t)s;
    }
    __syncthreads();

    // h = xsum @ Wn + edge_feats  (K=128 -> 4 WMMA per 16-col group)
#pragma unroll
    for (int g = 0; g < 4; ++g) {
        v8f acc = zero_v8f();
#pragma unroll
        for (int kk = 0; kk < 4; ++kk) {
            v16h a = load_A(&xsum[wv][0], D_NODE, kk * 32, lane);
            v16h b = load_B_frag(WnF, kk * 4 + g, lane);
            acc = wmma_f16(a, b, acc);
        }
#pragma unroll
        for (int v = 0; v < 8; ++v) {
            int r = v + 8 * hi;
            float hval = acc[v] + edge_feats[(size_t)(e0 + r) * D_EDGE + g * 16 + col];
            hbuf[wv][r * D_EDGE + g * 16 + col] = (half_t)hval;
        }
    }

    // q/k/v projections; softmax scale 1/sqrt(16) folded into q.
    const float* Ws[3] = {Wq, Wk, Wv};
    for (int mtx = 0; mtx < 3; ++mtx) {
        __syncthreads();
        stage_B_swizzled(WF, Ws[mtx], D_EDGE, 4, 8 * 512, tid, 128);
        __syncthreads();
        const float scale = (mtx == 0) ? 0.25f : 1.0f;
#pragma unroll
        for (int g = 0; g < 4; ++g) {
            v8f acc = zero_v8f();
#pragma unroll
            for (int kk = 0; kk < 2; ++kk) {
                v16h a = load_A(&hbuf[wv][0], D_EDGE, kk * 32, lane);
                v16h b = load_B_frag(WF, kk * 4 + g, lane);
                acc = wmma_f16(a, b, acc);
            }
            if (mtx == 2) {
                // transposed store: vT[dim][edge]; rows contiguous per lane
                half_t tmp[8];
#pragma unroll
                for (int v = 0; v < 8; ++v) tmp[v] = (half_t)acc[v];
                *(v8h*)(vtbuf + (size_t)(g * 16 + col) * E_EDGES + e0 + hi * 8) =
                    *(const v8h*)tmp;
            } else {
                half_t* dstb = (mtx == 0) ? qbuf : kbuf;
#pragma unroll
                for (int v = 0; v < 8; ++v) {
                    int r = v + 8 * hi;
                    dstb[(size_t)(e0 + r) * D_EDGE + g * 16 + col] =
                        (half_t)(acc[v] * scale);
                }
            }
        }
    }
}

// ---------------------------------------------------------------------------
// Stage one 64-key chunk (K rows, V^T rows, adj rows) into an LDS buffer.
// Async path issues global_load_async_to_lds_b128 (ASYNCcnt); caller waits.
// ---------------------------------------------------------------------------
__device__ __forceinline__ void stage_chunk(
    half_t* Kdst, half_t* Vdst, unsigned char* Adst,
    const half_t* __restrict__ kbuf, const half_t* __restrict__ vtbuf,
    const unsigned char* __restrict__ adj, int kc, int q0, int tid)
{
#if HAVE_ASYNC_LDS
    for (int i = tid; i < 512; i += 256)                       // K: 8 KB contiguous
        async_copy16(kbuf + (size_t)kc * D_EDGE + i * 8, Kdst + i * 8);
    for (int i = tid; i < 512; i += 256) {                     // V^T: 64 rows x 128 B
        int d = i >> 3, s = i & 7;
        async_copy16(vtbuf + (size_t)d * E_EDGES + kc + s * 8, Vdst + d * 64 + s * 8);
    }
    if (tid < 128) {                                           // adj: 32 rows x 64 B
        int r = tid >> 2, s = tid & 3;
        async_copy16(adj + (size_t)(q0 + r) * E_EDGES + kc + s * 16, Adst + r * 64 + s * 16);
    }
#else
    for (int i = tid; i < 512; i += 256)
        *(v8h*)(Kdst + i * 8) = *(const v8h*)(kbuf + (size_t)kc * D_EDGE + i * 8);
    for (int i = tid; i < 512; i += 256) {
        int d = i >> 3, s = i & 7;
        *(v8h*)(Vdst + d * 64 + s * 8) = *(const v8h*)(vtbuf + (size_t)d * E_EDGES + kc + s * 8);
    }
    for (int i = tid; i < 512; i += 256) {
        int r = i >> 4, c = i & 15;
        ((unsigned int*)Adst)[r * 16 + c] =
            *(const unsigned int*)(adj + (size_t)(q0 + r) * E_EDGES + kc + c * 4);
    }
#endif
}

// ---------------------------------------------------------------------------
// Kernel 2: masked flash-attention over all edges, double-buffered pipeline.
// 256 threads = 8 waves = HEADS(4) x query-subtiles(2); 32 queries per block.
// adj chunk staged once per block, shared by all 4 heads. grid = E/32 = 256.
// ---------------------------------------------------------------------------
__global__ void __launch_bounds__(256) attn_kernel(
    const half_t* __restrict__ qbuf, const half_t* __restrict__ kbuf,
    const half_t* __restrict__ vtbuf, const unsigned char* __restrict__ adj,
    half_t* __restrict__ ctx)
{
    __shared__ __align__(32) half_t Qs[32 * D_EDGE];            //  4 KB
    __shared__ __align__(32) half_t Ks[2][64 * D_EDGE];         // 16 KB  [key][dim]
    __shared__ __align__(32) half_t VsT[2][D_EDGE * 64];        // 16 KB  [dim][key]
    __shared__ __align__(32) unsigned char As[2][32 * 64];      //  4 KB
    __shared__ __align__(32) half_t Ps[8][16 * 32];             //  8 KB per-wave P scratch

    const int tid  = threadIdx.x;
    const int lane = tid & 31;
    const int wv   = tid >> 5;
    const int head = wv & 3;
    const int qsub = wv >> 2;
    const int q0   = blockIdx.x * 32;
    const int hd0  = head * HD;
    const int col  = lane & 15, hi = lane >> 4;

    for (int i = tid; i < 32 * D_EDGE / 8; i += 256)
        ((v8h*)Qs)[i] = ((const v8h*)(qbuf + (size_t)q0 * D_EDGE))[i];

    // prologue: stage chunk 0 into buffer 0
    stage_chunk(Ks[0], VsT[0], As[0], kbuf, vtbuf, adj, 0, q0, tid);
#if HAVE_ASYNC_LDS
    async_wait0();
#endif
    __syncthreads();

    // Q fragment (head slice, K padded 16->32), reused across the whole key loop.
    const v16h aq = load_A_k16(&Qs[(qsub * 16) * D_EDGE + hd0], D_EDGE, lane);

    float mrun[8], lrun[8];
    v8f acc = zero_v8f();
#pragma unroll
    for (int v = 0; v < 8; ++v) { mrun[v] = -1e30f; lrun[v] = 0.0f; }

    int cur = 0;
    for (int kc = 0; kc < E_EDGES; kc += 64) {
        // issue DMA of the NEXT chunk into the idle buffer; overlaps compute
        if (kc + 64 < E_EDGES)
            stage_chunk(Ks[cur ^ 1], VsT[cur ^ 1], As[cur ^ 1],
                        kbuf, vtbuf, adj, kc + 64, q0, tid);

        const half_t* Kc = Ks[cur];
        const half_t* Vc = VsT[cur];
        const unsigned char* Ac = As[cur];

#pragma unroll
        for (int p = 0; p < 2; ++p) {            // two 32-key groups per chunk
            v8f s0 = wmma_f16(aq, load_B_kT16(Kc + (p * 32 +  0) * D_EDGE + hd0, D_EDGE, lane), zero_v8f());
            v8f s1 = wmma_f16(aq, load_B_kT16(Kc + (p * 32 + 16) * D_EDGE + hd0, D_EDGE, lane), zero_v8f());

            // mask (reference: where(adj, s, -1e9) then softmax over all keys)
#pragma unroll
            for (int v = 0; v < 8; ++v) {
                int r = qsub * 16 + v + 8 * hi;
                s0[v] = Ac[r * 64 + p * 32 + col]      ? s0[v] : -1e9f;
                s1[v] = Ac[r * 64 + p * 32 + 16 + col] ? s1[v] : -1e9f;
            }
            // online softmax per row group; write P scratch in f16
#pragma unroll
            for (int v = 0; v < 8; ++v) {
                float tmax = red_max16(fmaxf(s0[v], s1[v]));
                float mnew = fmaxf(mrun[v], tmax);
                float ef   = __expf(mrun[v] - mnew);
                float p0   = __expf(s0[v] - mnew);
                float p1   = __expf(s1[v] - mnew);
                lrun[v] = lrun[v] * ef + red_sum16(p0 + p1);
                mrun[v] = mnew;
                acc[v] *= ef;
                int r = v + 8 * hi;
                Ps[wv][r * 32 + col]      = (half_t)p0;
                Ps[wv][r * 32 + 16 + col] = (half_t)p1;
            }
            // acc += P @ V  (K=32 keys; same-wave DS ops are in-order)
            v16h ap = load_A(&Ps[wv][0], 32, 0, lane);
            v16h bv = load_B_vT(Vc, p * 32, hd0, lane);
            acc = wmma_f16(ap, bv, acc);
        }

        // next chunk landed (own ASYNCcnt) + everyone done reading `cur`
#if HAVE_ASYNC_LDS
        async_wait0();
#endif
        __syncthreads();
        cur ^= 1;
    }

    // normalize and emit ctx slice for this head
#pragma unroll
    for (int v = 0; v < 8; ++v) {
        int r = v + 8 * hi;
        float o = acc[v] / lrun[v];
        ctx[(size_t)(q0 + qsub * 16 + r) * D_EDGE + hd0 + col] = (half_t)o;
    }
}

// ---------------------------------------------------------------------------
// Kernel 3: edge_out = ctx@Wo; out = gelu(edge_out@W1+b1)@W2 + b2 (f32 out).
// 128 threads = 4 waves, one 16-edge tile per wave. grid = E/64 = 128.
// ---------------------------------------------------------------------------
__device__ __forceinline__ float gelu_tanh(float x) {
    float x3 = x * x * x;
    return 0.5f * x * (1.0f + tanhf(0.7978845608028654f * (x + 0.044715f * x3)));
}

__global__ void __launch_bounds__(128) out_kernel(
    const half_t* __restrict__ ctx, const float* __restrict__ Wo,
    const float* __restrict__ W1, const float* __restrict__ b1,
    const float* __restrict__ W2, const float* __restrict__ b2,
    float* __restrict__ out)
{
    __shared__ __align__(32) half_t WoF[8 * 512];          // 8 KB, 2x4 frags
    __shared__ __align__(32) half_t W1F[4 * 512];          // 4 KB, 2x2 frags
    __shared__ __align__(32) half_t W2F[2 * 512];          // 2 KB, 1x2 frags
    __shared__ float b1L[EMBED], b2L[EMBED];
    __shared__ __align__(32) half_t eo[4][16 * D_EDGE];    // 8 KB
    __shared__ __align__(32) half_t xb[4][16 * EMBED];     // 4 KB

    const int tid  = threadIdx.x;
    const int lane = tid & 31;
    const int wv   = tid >> 5;
    const int e0   = (blockIdx.x * 4 + wv) * 16;
    const int col  = lane & 15, hi = lane >> 4;

    stage_B_swizzled(WoF, Wo, D_EDGE, 4, 8 * 512, tid, 128);
    stage_B_swizzled(W1F, W1, EMBED, 2, 4 * 512, tid, 128);
    stage_B_swizzled(W2F, W2, EMBED, 2, 2 * 512, tid, 128);
    if (tid < EMBED) { b1L[tid] = b1[tid]; b2L[tid] = b2[tid]; }
    __syncthreads();

    // edge_out = ctx @ Wo
#pragma unroll
    for (int g = 0; g < 4; ++g) {
        v8f acc = zero_v8f();
#pragma unroll
        for (int kk = 0; kk < 2; ++kk) {
            v16h a = load_A(ctx + (size_t)e0 * D_EDGE, D_EDGE, kk * 32, lane);
            v16h b = load_B_frag(WoF, kk * 4 + g, lane);
            acc = wmma_f16(a, b, acc);
        }
#pragma unroll
        for (int v = 0; v < 8; ++v) {
            int r = v + 8 * hi;
            eo[wv][r * D_EDGE + g * 16 + col] = (half_t)acc[v];
        }
    }

    // x = gelu(edge_out @ W1 + b1)
#pragma unroll
    for (int g = 0; g < 2; ++g) {
        v8f acc = zero_v8f();
#pragma unroll
        for (int kk = 0; kk < 2; ++kk) {
            v16h a = load_A(&eo[wv][0], D_EDGE, kk * 32, lane);
            v16h b = load_B_frag(W1F, kk * 2 + g, lane);
            acc = wmma_f16(a, b, acc);
        }
#pragma unroll
        for (int v = 0; v < 8; ++v) {
            int r = v + 8 * hi;
            float x = acc[v] + b1L[g * 16 + col];
            xb[wv][r * EMBED + g * 16 + col] = (half_t)gelu_tanh(x);
        }
    }

    // out = x @ W2 + b2   (K=32 exactly -> single WMMA per column group)
#pragma unroll
    for (int g = 0; g < 2; ++g) {
        v16h a = load_A(&xb[wv][0], EMBED, 0, lane);
        v16h b = load_B_frag(W2F, g, lane);
        v8f acc = wmma_f16(a, b, zero_v8f());
#pragma unroll
        for (int v = 0; v < 8; ++v) {
            int r = v + 8 * hi;
            out[(size_t)(e0 + r) * EMBED + g * 16 + col] = acc[v] + b2L[g * 16 + col];
        }
    }
}

// ---------------------------------------------------------------------------
extern "C" void kernel_launch(void* const* d_in, const int* in_sizes, int n_in,
                              void* d_out, int out_size, void* d_ws, size_t ws_size,
                              hipStream_t stream) {
    (void)in_sizes; (void)n_in; (void)out_size; (void)ws_size;
    const float* edge_feats = (const float*)d_in[0];
    const float* node_feats = (const float*)d_in[1];
    const float* Wn  = (const float*)d_in[2];
    const float* Wq  = (const float*)d_in[3];
    const float* Wk  = (const float*)d_in[4];
    const float* Wv  = (const float*)d_in[5];
    const float* Wo  = (const float*)d_in[6];
    const float* W1  = (const float*)d_in[7];
    const float* b1  = (const float*)d_in[8];
    const float* W2  = (const float*)d_in[9];
    const float* b2  = (const float*)d_in[10];
    const int*   edge_index = (const int*)d_in[11];
    const unsigned char* edge_adj = (const unsigned char*)d_in[12];
    float* out = (float*)d_out;

    // workspace: f16 Q, K (row-major [E][64]), V^T ([64][E]), CTX  (4 x 1 MB)
    half_t* qbuf  = (half_t*)d_ws;
    half_t* kbuf  = qbuf  + (size_t)E_EDGES * D_EDGE;
    half_t* vtbuf = kbuf  + (size_t)E_EDGES * D_EDGE;
    half_t* ctxb  = vtbuf + (size_t)E_EDGES * D_EDGE;

    prep_kernel<<<E_EDGES / 64, 128, 0, stream>>>(edge_feats, node_feats, Wn, Wq, Wk, Wv,
                                                  edge_index, qbuf, kbuf, vtbuf);
    attn_kernel<<<E_EDGES / 32, 256, 0, stream>>>(qbuf, kbuf, vtbuf, edge_adj, ctxb);
    out_kernel<<<E_EDGES / 64, 128, 0, stream>>>(ctxb, Wo, W1, b1, W2, b2, out);
}
